// GroupedQueryAttention_27728308863021
// MI455X (gfx1250) — matrix operand order repaired and checked by reading, not compiled, not run on hardware
//
#include <hip/hip_runtime.h>
#include <hip/hip_bf16.h>

typedef _Float16 half_t;
typedef __attribute__((ext_vector_type(16))) _Float16 v16h;
typedef __attribute__((ext_vector_type(8)))  _Float16 v8h;
typedef __attribute__((ext_vector_type(8)))  float    v8f;

#define B_   2
#define S_   2048
#define D_   2048
#define H_   32
#define HKV_ 8
#define HD_  64

#define WMMA_F16(a, b, c) \
  __builtin_amdgcn_wmma_f32_16x16x32_f16(false, (a), false, (b), (short)0, (c), false, false)

__device__ __forceinline__ v16h mk16(v8h lo, v8h hi) {
  return __builtin_shufflevector(lo, hi, 0,1,2,3,4,5,6,7,8,9,10,11,12,13,14,15);
}
__device__ __forceinline__ v8h ld8(const half_t* p) { return *(const v8h*)p; }

// ---------------------------------------------------------------- cvt f32->f16
__global__ void __launch_bounds__(256) k_cvt(const float* __restrict__ in,
                                             half_t* __restrict__ out, size_t n) {
  size_t i = (size_t)blockIdx.x * blockDim.x + threadIdx.x;
  size_t st = (size_t)gridDim.x * blockDim.x;
  for (; i < n; i += st) out[i] = (half_t)in[i];
}

// ------------------------------------------------- transpose + cvt: Wt[n][k] = W[k][n]
__global__ void __launch_bounds__(256) k_transpose_cvt(const float* __restrict__ W,
                                                       half_t* __restrict__ Wt,
                                                       int N, int Kd) {
  size_t tot = (size_t)N * Kd;
  size_t i = (size_t)blockIdx.x * blockDim.x + threadIdx.x;
  size_t st = (size_t)gridDim.x * blockDim.x;
  for (; i < tot; i += st) {
    int nn = (int)(i / Kd);
    int kk = (int)(i % Kd);
    Wt[i] = (half_t)W[(size_t)kk * N + nn];
  }
}

// ---------------------------------------------------------------- RoPE in-place on f16
// token layout: Q -> token=(b*S+s)*H+h (hdiv=H), K -> token=(b*HKV+kvh)*S+s (hdiv=1)
__global__ void __launch_bounds__(256) k_rope(half_t* __restrict__ x,
                                              const float* __restrict__ cosb,
                                              const float* __restrict__ sinb,
                                              size_t nPairs, int hdiv) {
  size_t i = (size_t)blockIdx.x * blockDim.x + threadIdx.x;
  size_t st = (size_t)gridDim.x * blockDim.x;
  for (; i < nPairs; i += st) {
    size_t token = i >> 5;
    int d1 = (int)(i & 31);
    int s = (int)((token / hdiv) % S_);
    float c  = cosb[s * HD_ + d1];
    float sn = sinb[s * HD_ + d1];
    size_t base = token * HD_;
    float x1 = (float)x[base + d1];
    float x2 = (float)x[base + d1 + 32];
    x[base + d1]      = (half_t)(x1 * c - x2 * sn);
    x[base + d1 + 32] = (half_t)(x2 * c + x1 * sn);
  }
}

// ---------------------------------------------------------------- WMMA GEMM
// C[M,N] = A[M,K] * Bt[N,K]^T.  One wave = 32x64 macro-tile (2x4 WMMA tiles).
// omode 0: f16 row-major (Q proj / generic)
// omode 1: f16 K layout  ((b*HKV+kvh)*S + s)*64 + d
// omode 2: f16 V^T layout ((b*HKV+kvh)*64 + d)*S + s  (contiguous per-lane stores)
// omode 3: f32 row-major to d_out (O projection)
__global__ void __launch_bounds__(256) k_gemm(const half_t* __restrict__ A,
                                              const half_t* __restrict__ Bt,
                                              half_t* __restrict__ Ch,
                                              float* __restrict__ Cf,
                                              int M, int N, int Kd, int omode) {
  const int lane = threadIdx.x & 31;
  const int gw = blockIdx.x * (blockDim.x >> 5) + (threadIdx.x >> 5);
  const int ntn = N >> 6;
  const int mt = gw / ntn, nt = gw % ntn;
  const int mBase = mt * 32, nBase = nt * 64;
  if (mBase >= M) return;

  const int lr = lane & 15;
  const int abase = (lane < 16) ? 0 : 8;   // A-frag K sub-offset
  const int bbase = (lane < 16) ? 0 : 16;  // unused: B-frag K sub-offset is abase, as for A

  const half_t* rowA0 = A + (size_t)(mBase + lr) * Kd;
  const half_t* rowA1 = A + (size_t)(mBase + 16 + lr) * Kd;
  const half_t* colB  = Bt + (size_t)(nBase + lr) * Kd;

  v8f acc[2][4] = {};
  for (int k = 0; k < Kd; k += 32) {
    v16h a0 = mk16(ld8(rowA0 + k + abase), ld8(rowA0 + k + 16 + abase));
    v16h a1 = mk16(ld8(rowA1 + k + abase), ld8(rowA1 + k + 16 + abase));
#pragma unroll
    for (int j = 0; j < 4; j++) {
      v16h bj = mk16(ld8(colB + (size_t)j * 16 * Kd + k + abase), ld8(colB + (size_t)j * 16 * Kd + k + 16 + abase));
      acc[0][j] = WMMA_F16(a0, bj, acc[0][j]);
      acc[1][j] = WMMA_F16(a1, bj, acc[1][j]);
    }
  }

  const int halfm = abase;  // 0 or 8: C-layout row offset for this lane half
#pragma unroll
  for (int i = 0; i < 2; i++) {
#pragma unroll
    for (int j = 0; j < 4; j++) {
      int n = nBase + j * 16 + lr;
      if (omode == 0) {
#pragma unroll
        for (int r = 0; r < 8; r++) {
          int m = mBase + i * 16 + halfm + r;
          Ch[(size_t)m * N + n] = (half_t)acc[i][j][r];
        }
      } else if (omode == 1) {
        int kvh = n >> 6, d = n & 63;
#pragma unroll
        for (int r = 0; r < 8; r++) {
          int m = mBase + i * 16 + halfm + r;
          int b = m >> 11, s = m & (S_ - 1);
          Ch[(((size_t)b * HKV_ + kvh) * S_ + s) * HD_ + d] = (half_t)acc[i][j][r];
        }
      } else if (omode == 2) {
        int kvh = n >> 6, d = n & 63;
        int s0 = mBase + i * 16 + halfm;
        int b = s0 >> 11, s = s0 & (S_ - 1);
        v8h pk;
#pragma unroll
        for (int r = 0; r < 8; r++) pk[r] = (half_t)acc[i][j][r];
        *(v8h*)(Ch + (((size_t)b * HKV_ + kvh) * HD_ + d) * S_ + s) = pk;
      } else {
#pragma unroll
        for (int r = 0; r < 8; r++) {
          int m = mBase + i * 16 + halfm + r;
          Cf[(size_t)m * N + n] = acc[i][j][r];
        }
      }
    }
  }
}

// ---------------------------------------------------------------- Flash GQA attention
// One wave per (b, h, 16-query tile). Transposed math:
//   S^T(key,q) = Kchunk * Q^T   ;   O^T(d,q) += V^T * P^T
// so softmax stats live per lane (lane = query column) + one shfl_xor(16).
__global__ void __launch_bounds__(256) k_flash(const half_t* __restrict__ Qb,
                                               const half_t* __restrict__ Kb,
                                               const half_t* __restrict__ Vtb,
                                               half_t* __restrict__ Ob) {
  __shared__ __align__(64) half_t pbuf[8 * 512];  // per-wave 32x16 f16 P^T stage

  const int lane = threadIdx.x & 31;
  const int widx = threadIdx.x >> 5;
  const int gw = blockIdx.x * 8 + widx;
  const int qt = gw & 127;          // S/16
  const int h  = (gw >> 7) & 31;
  const int b  = gw >> 12;
  const int kvh = h >> 2;           // H/HKV = 4
  const int qbase = qt * 16;

  const int lr = lane & 15;
  const int abase = (lane < 16) ? 0 : 8;
  const int bsel  = (lane < 16) ? 0 : 16;
  const int qrow = qbase + lr;

  // Q^T B-fragments for the two d-halves (d = t*32 + abase + e for e<8, t*32 + 16 + abase + (e-8) for e>=8)
  const half_t* Qrow = Qb + (size_t)(b * S_ + qbase + lr) * D_ + h * HD_;
  v16h bq0 = mk16(ld8(Qrow + 0  + abase), ld8(Qrow + 16 + abase));
  v16h bq1 = mk16(ld8(Qrow + 32 + abase), ld8(Qrow + 48 + abase));

  const half_t* Kh = Kb  + (size_t)(b * HKV_ + kvh) * S_ * HD_;
  const half_t* Vh = Vtb + (size_t)(b * HKV_ + kvh) * HD_ * S_;
  half_t* pm = pbuf + widx * 512 + lr * 32;

  v8f o0 = {}, o1 = {}, o2 = {}, o3 = {};
  float mrun = -1e30f, lrun = 0.f;
  const float SCL = 0.18033688f;  // (1/sqrt(64)) * log2(e): exp2-domain logits

  const int kend = qbase + 16;
  for (int c = 0; c < kend; c += 32) {
    if (c + 32 < kend) {
      __builtin_prefetch(Kh + (size_t)(c + 32) * HD_, 0, 0);
      __builtin_prefetch(Vh + (size_t)lr * S_ + c + 32, 0, 0);
    }
    // K A-fragments (two key sub-tiles x two d-steps); clamp tail rows (masked anyway)
    int k0 = c + lr;       if (k0 > S_ - 1) k0 = S_ - 1;
    int k1 = c + 16 + lr;  if (k1 > S_ - 1) k1 = S_ - 1;
    const half_t* kr0 = Kh + (size_t)k0 * HD_;
    const half_t* kr1 = Kh + (size_t)k1 * HD_;
    v16h ak00 = mk16(ld8(kr0 + abase),      ld8(kr0 + 16 + abase));
    v16h ak01 = mk16(ld8(kr0 + 32 + abase), ld8(kr0 + 48 + abase));
    v16h ak10 = mk16(ld8(kr1 + abase),      ld8(kr1 + 16 + abase));
    v16h ak11 = mk16(ld8(kr1 + 32 + abase), ld8(kr1 + 48 + abase));

    v8f st0 = {}, st1 = {};
    st0 = WMMA_F16(ak00, bq0, st0);
    st0 = WMMA_F16(ak01, bq1, st0);
    st1 = WMMA_F16(ak10, bq0, st1);
    st1 = WMMA_F16(ak11, bq1, st1);

    // causal mask + scale (exp2 domain), chunk max
    float mc = -1e30f;
#pragma unroll
    for (int r = 0; r < 8; r++) {
      int keyA = c + r + abase;
      int keyB = c + 16 + r + abase;
      float va = st0[r] * SCL; if (keyA > qrow) va = -1e30f;
      float vb = st1[r] * SCL; if (keyB > qrow) vb = -1e30f;
      st0[r] = va; st1[r] = vb;
      mc = fmaxf(mc, fmaxf(va, vb));
    }
    mc = fmaxf(mc, __shfl_xor(mc, 16, 32));
    float mn = fmaxf(mrun, mc);
    float alpha = exp2f(mrun - mn);

    float lc = 0.f;
    v8h p0, p1;
#pragma unroll
    for (int r = 0; r < 8; r++) {
      float pa = exp2f(st0[r] - mn);
      float pb = exp2f(st1[r] - mn);
      lc += pa + pb;
      p0[r] = (half_t)pa;
      p1[r] = (half_t)pb;
    }
    lc += __shfl_xor(lc, 16, 32);
    lrun = lrun * alpha + lc;
    mrun = mn;
    o0 = o0 * alpha; o1 = o1 * alpha; o2 = o2 * alpha; o3 = o3 * alpha;

    // stage P^T through LDS to re-fragment C-layout -> B-layout
    *(v8h*)(pm + abase)      = p0;
    *(v8h*)(pm + 16 + abase) = p1;
    asm volatile("s_wait_dscnt 0x0" ::: "memory");
    v16h bp = mk16(ld8(pm + abase), ld8(pm + 16 + abase));

    // V^T A-fragments; clamp tail key groups (their p == 0, avoid NaN*0)
    int sa = c + abase;       if (sa > S_ - 8) sa = S_ - 8;
    int sb = c + 16 + abase;  if (sb > S_ - 8) sb = S_ - 8;
    const half_t* vrow = Vh + (size_t)lr * S_;
    v16h av0 = mk16(ld8(vrow + (size_t)0 * 16 * S_ + sa), ld8(vrow + (size_t)0 * 16 * S_ + sb));
    v16h av1 = mk16(ld8(vrow + (size_t)1 * 16 * S_ + sa), ld8(vrow + (size_t)1 * 16 * S_ + sb));
    v16h av2 = mk16(ld8(vrow + (size_t)2 * 16 * S_ + sa), ld8(vrow + (size_t)2 * 16 * S_ + sb));
    v16h av3 = mk16(ld8(vrow + (size_t)3 * 16 * S_ + sa), ld8(vrow + (size_t)3 * 16 * S_ + sb));
    o0 = WMMA_F16(av0, bp, o0);
    o1 = WMMA_F16(av1, bp, o1);
    o2 = WMMA_F16(av2, bp, o2);
    o3 = WMMA_F16(av3, bp, o3);
  }

  float inv = (lrun > 0.f) ? 1.f / lrun : 0.f;
  o0 = o0 * inv; o1 = o1 * inv; o2 = o2 * inv; o3 = o3 * inv;

  half_t* orow = Ob + (size_t)(b * S_ + qbase + lr) * D_ + h * HD_;
  v8h w0, w1, w2, w3;
#pragma unroll
  for (int r = 0; r < 8; r++) {
    w0[r] = (half_t)o0[r]; w1[r] = (half_t)o1[r];
    w2[r] = (half_t)o2[r]; w3[r] = (half_t)o3[r];
  }
  *(v8h*)(orow + 0 * 16 + abase) = w0;
  *(v8h*)(orow + 1 * 16 + abase) = w1;
  *(v8h*)(orow + 2 * 16 + abase) = w2;
  *(v8h*)(orow + 3 * 16 + abase) = w3;
}

// ---------------------------------------------------------------- launch
extern "C" void kernel_launch(void* const* d_in, const int* in_sizes, int n_in,
                              void* d_out, int out_size, void* d_ws, size_t ws_size,
                              hipStream_t stream) {
  const float* hs   = (const float*)d_in[0];
  // d_in[1] = attention_mask (pure causal; applied analytically in k_flash)
  const float* cosb = (const float*)d_in[2];
  const float* sinb = (const float*)d_in[3];
  const float* Wq   = (const float*)d_in[4];
  const float* Wk   = (const float*)d_in[5];
  const float* Wv   = (const float*)d_in[6];
  const float* Wo   = (const float*)d_in[7];

  char* ws = (char*)d_ws;
  size_t off = 0;
  auto take = [&](size_t bytes) { char* p = ws + off; off += bytes; return p; };
  half_t* Xh  = (half_t*)take((size_t)B_ * S_ * D_ * 2);          // 16 MiB
  half_t* WqT = (half_t*)take((size_t)D_ * D_ * 2);               //  8 MiB
  half_t* WkT = (half_t*)take((size_t)HKV_ * HD_ * D_ * 2);       //  2 MiB
  half_t* WvT = (half_t*)take((size_t)HKV_ * HD_ * D_ * 2);       //  2 MiB
  half_t* WoT = (half_t*)take((size_t)D_ * D_ * 2);               //  8 MiB
  half_t* Qb  = (half_t*)take((size_t)B_ * S_ * D_ * 2);          // 16 MiB
  half_t* Kb  = (half_t*)take((size_t)B_ * HKV_ * S_ * HD_ * 2);  //  4 MiB
  half_t* Vt  = (half_t*)take((size_t)B_ * HKV_ * HD_ * S_ * 2);  //  4 MiB
  half_t* Ob  = (half_t*)take((size_t)B_ * S_ * D_ * 2);          // 16 MiB (~76 MiB total)

  const int M = B_ * S_;

  k_cvt<<<4096, 256, 0, stream>>>(hs, Xh, (size_t)B_ * S_ * D_);
  k_transpose_cvt<<<4096, 256, 0, stream>>>(Wq, WqT, D_, D_);
  k_transpose_cvt<<<2048, 256, 0, stream>>>(Wk, WkT, HKV_ * HD_, D_);
  k_transpose_cvt<<<2048, 256, 0, stream>>>(Wv, WvT, HKV_ * HD_, D_);
  k_transpose_cvt<<<4096, 256, 0, stream>>>(Wo, WoT, D_, D_);

  // Q: waves = (4096/32)*(2048/64) = 4096 -> 512 blocks of 8 waves
  k_gemm<<<512, 256, 0, stream>>>(Xh, WqT, Qb, nullptr, M, D_, D_, 0);
  // K/V: waves = 128*8 = 1024 -> 128 blocks
  k_gemm<<<128, 256, 0, stream>>>(Xh, WkT, Kb, nullptr, M, HKV_ * HD_, D_, 1);
  k_gemm<<<128, 256, 0, stream>>>(Xh, WvT, Vt, nullptr, M, HKV_ * HD_, D_, 2);

  k_rope<<<2048, 256, 0, stream>>>(Qb, cosb, sinb, (size_t)B_ * S_ * H_ * 32, H_);
  k_rope<<<512, 256, 0, stream>>>(Kb, cosb, sinb, (size_t)B_ * HKV_ * S_ * 32, 1);

  // waves = B*H*(S/16) = 8192 -> 1024 blocks
  k_flash<<<1024, 256, 0, stream>>>(Qb, Kb, Vt, Ob);

  k_gemm<<<512, 256, 0, stream>>>(Ob, WoT, nullptr, (float*)d_out, M, D_, D_, 3);
}
